// W8A8BFP16FP16Conv1D_32968168964174
// MI455X (gfx1250) — 8-bit matrix operand order repaired and checked by reading, compiled, not run on hardware
//
#include <hip/hip_runtime.h>
#include <hip/hip_fp16.h>
#include <stdint.h>

#define M_DIM 8192
#define N_DIM 4096
#define K_DIM 4096

#define BM 128
#define BN 128
#define BK 64
#define NKSTEPS (K_DIM / BK)
#define KB      (K_DIM / 64)
#define SLAB    (BM * BK)

typedef __attribute__((ext_vector_type(8))) int v8i;
typedef int b128_t __attribute__((vector_size(16)));

#ifndef __has_builtin
#define __has_builtin(x) 0
#endif

#if __has_builtin(__builtin_amdgcn_global_load_async_to_lds_b128)
#define USE_ASYNC 1
#else
#define USE_ASYNC 0
#endif

typedef __attribute__((address_space(1))) b128_t* g_b128_p;
typedef __attribute__((address_space(3))) b128_t* l_b128_p;

__device__ __forceinline__ void copy16(const char* g, char* l) {
#if USE_ASYNC
  __builtin_amdgcn_global_load_async_to_lds_b128(
      (g_b128_p)(uintptr_t)g, (l_b128_p)(unsigned)(uintptr_t)l, 0, 0);
#else
  *(int4*)l = *(const int4*)g;
#endif
}

__device__ __forceinline__ void wait_older_slab() {
#if USE_ASYNC
  asm volatile("s_wait_asynccnt 4" ::: "memory");
#endif
}
__device__ __forceinline__ void wait_all_slabs() {
#if USE_ASYNC
  asm volatile("s_wait_asynccnt 0" ::: "memory");
#endif
}

__device__ __forceinline__ unsigned pack4(int a, int b, int c, int d) {
  return (a & 0xff) | ((b & 0xff) << 8) | ((c & 0xff) << 16) | (((unsigned)d) << 24);
}


__global__ void pack_x_kernel(const int* __restrict__ x,
                              char* __restrict__ a8f) {
  long t = (long)blockIdx.x * blockDim.x + threadIdx.x;
  int c = (int)(t & (K_DIM / 16 - 1));
  int m = (int)(t >> 8);
  if (m >= M_DIM) return;

  const int4* src = (const int4*)(x + (long)m * K_DIM + c * 16);
  int4 w0 = src[0], w1 = src[1], w2 = src[2], w3 = src[3];
  unsigned lo0 = pack4(w0.x, w0.y, w0.z, w0.w);
  unsigned lo1 = pack4(w1.x, w1.y, w1.z, w1.w);
  unsigned hi0 = pack4(w2.x, w2.y, w2.z, w2.w);
  unsigned hi1 = pack4(w3.x, w3.y, w3.z, w3.w);

  int tr  = m >> 4, l15 = m & 15;
  int ks  = c >> 2;
  int j   = c & 3;
  long base = ((long)(tr * KB + ks)) << 10;
  uint2 lo; lo.x = lo0; lo.y = lo1;
  uint2 hi; hi.x = hi0; hi.y = hi1;
  *(uint2*)(a8f + base + ((j & 1) * 16 + l15) * 32 + (j >> 1) * 16) = lo;
  *(uint2*)(a8f + base + ((j & 1) * 16 + l15) * 32 + (j >> 1) * 16 + 8) = hi;
}

__global__ void pack_w_kernel(const int* __restrict__ w,
                              char* __restrict__ b8f) {
  long t = (long)blockIdx.x * blockDim.x + threadIdx.x;
  int n = (int)(t & (N_DIM - 1));
  int c = (int)(t >> 12);
  if (c >= K_DIM / 16) return;

  const int* col = w + n;
  unsigned b[4];
#pragma unroll
  for (int i = 0; i < 4; ++i) {
    int e0 = col[(long)(c * 16 + i * 4 + 0) * N_DIM];
    int e1 = col[(long)(c * 16 + i * 4 + 1) * N_DIM];
    int e2 = col[(long)(c * 16 + i * 4 + 2) * N_DIM];
    int e3 = col[(long)(c * 16 + i * 4 + 3) * N_DIM];
    b[i] = pack4(e0, e1, e2, e3);
  }
  int ks = c >> 2, cc = c & 3;
  int kh = cc & 1, half = cc >> 1;
  int lanev = (kh << 4) | (n & 15);
  long base = ((long)((n >> 4) * KB + ks)) << 10;
  uint4 q; q.x = b[0]; q.y = b[1]; q.z = b[2]; q.w = b[3];
  *(uint4*)(b8f + base + lanev * 32 + half * 16) = q;
}


__global__ void __launch_bounds__(256)
gemm_i8_kernel(const char* __restrict__ A8F,
               const char* __restrict__ B8F,
               const __half* __restrict__ bias,
               const float* __restrict__ alphaPtr,
               float* __restrict__ out) {
  __shared__ __align__(32) char Asm[2][SLAB];
  __shared__ __align__(32) char Bsm[2][SLAB];

  const int tid  = threadIdx.x;
  const int lane = tid & 31;
  const int wid  = tid >> 5;
  const int waveM = wid >> 2;
  const int waveN = wid & 3;
  const int l15 = lane & 15;
  const int kh  = lane >> 4;

  const int bmTr = blockIdx.y * (BM / 16);
  const int bnTr = blockIdx.x * (BN / 16);

  char* const aBase = &Asm[0][0];
  char* const bBase = &Bsm[0][0];

  v8i acc[4][2];
#pragma unroll
  for (int mi = 0; mi < 4; ++mi)
#pragma unroll
    for (int ni = 0; ni < 2; ++ni)
#pragma unroll
      for (int j = 0; j < 8; ++j) acc[mi][ni][j] = 0;

  const char* gA[2];
  const char* gB[2];
  int slot[2];
#pragma unroll
  for (int c = 0; c < 2; ++c) {
    int o      = (tid + c * 256) * 16;
    int blk    = o >> 10;
    int within = o & 1023;
    slot[c] = o;
    gA[c] = A8F + (((long)(bmTr + blk) * KB) << 10) + within;
    gB[c] = B8F + (((long)(bnTr + blk) * KB) << 10) + within;
  }

#pragma unroll
  for (int c = 0; c < 2; ++c) {
    copy16(gA[c], aBase + slot[c]);
    copy16(gB[c], bBase + slot[c]);
    gA[c] += 1024; gB[c] += 1024;
  }
#pragma unroll
  for (int c = 0; c < 2; ++c) {
    copy16(gA[c], aBase + SLAB + slot[c]);
    copy16(gB[c], bBase + SLAB + slot[c]);
    gA[c] += 1024; gB[c] += 1024;
  }

  unsigned ldsOff = 0;

#pragma unroll 1
  for (int ks = 0; ks < NKSTEPS; ++ks) {
    if (ks == NKSTEPS - 1) wait_all_slabs();
    else                   wait_older_slab();
    __syncthreads();

    char* const af = aBase + ldsOff;
    char* const bf = bBase + ldsOff;

    v8i afrag[4];
#pragma unroll
    for (int mi = 0; mi < 4; ++mi)
      afrag[mi] = *(const v8i*)(af + ((waveM * 4 + mi) << 10) + lane * 32);

    v8i bfrag[2];
#pragma unroll
    for (int ni = 0; ni < 2; ++ni)
      bfrag[ni] = *(const v8i*)(bf + ((waveN * 2 + ni) << 10) + lane * 32);

    asm volatile("s_wait_dscnt 0" ::: "memory");
    __syncthreads();

    if (ks + 2 < NKSTEPS) {
#pragma unroll
      for (int c = 0; c < 2; ++c) {
        copy16(gA[c], af + slot[c]);
        copy16(gB[c], bf + slot[c]);
        gA[c] += 1024; gB[c] += 1024;
      }
    }

#pragma unroll
    for (int mi = 0; mi < 4; ++mi)
#pragma unroll
      for (int ni = 0; ni < 2; ++ni)
        acc[mi][ni] = __builtin_amdgcn_wmma_i32_16x16x64_iu8(
             true, afrag[mi],  true, bfrag[ni],
            acc[mi][ni],  false,  false);

    ldsOff ^= SLAB;
  }

  const float alpha = *alphaPtr;
  const int bm = bmTr * 16, bn = bnTr * 16;
#pragma unroll
  for (int ni = 0; ni < 2; ++ni) {
    const int col = bn + waveN * 32 + ni * 16 + l15;
    const float bc = __half2float(bias[col]);
#pragma unroll
    for (int mi = 0; mi < 4; ++mi) {
      v8i c = acc[mi][ni];
      const int rowb = bm + waveM * 64 + mi * 16 + kh * 8;
#pragma unroll
      for (int j = 0; j < 8; ++j)
        out[(long)(rowb + j) * N_DIM + col] = ((float)c[j] + bc) * alpha;
    }
  }
}


extern "C" void kernel_launch(void* const* d_in, const int* in_sizes, int n_in,
                              void* d_out, int out_size, void* d_ws, size_t ws_size,
                              hipStream_t stream) {
  const int*    x     = (const int*)d_in[0];
  const int*    w     = (const int*)d_in[1];
  const __half* bias  = (const __half*)d_in[2];
  const float*  alpha = (const float*)d_in[3];
  float*        out   = (float*)d_out;

  char* A8F = (char*)d_ws;
  char* B8F = (char*)d_ws + (size_t)M_DIM * K_DIM;

  {
    long nt = (long)M_DIM * K_DIM / 16;
    pack_x_kernel<<<(int)(nt / 256), 256, 0, stream>>>(x, A8F);
  }
  {
    long nt = (long)N_DIM * K_DIM / 16;
    pack_w_kernel<<<(int)(nt / 256), 256, 0, stream>>>(w, B8F);
  }
  dim3 grid(N_DIM / BN, M_DIM / BM);
  gemm_i8_kernel<<<grid, 256, 0, stream>>>(A8F, B8F, bias, alpha, out);
}
